// drug_2d_encoder_17205638988649
// MI455X (gfx1250) — matrix operand order repaired and checked by reading, not compiled, not run on hardware
//
#include <hip/hip_runtime.h>
#include <hip/hip_bf16.h>

typedef __attribute__((ext_vector_type(16))) __bf16 v16bf;
typedef __attribute__((ext_vector_type(8)))  float  v8f;

#define D        128
#define LAYERS   5
#define ROWS     64            // rows per block in MLP kernel
#define AS_STRIDE 136          // 128 + 8 pad (bf16 elems, multiple of 8)
#define YS_STRIDE 264          // 256 + 8 pad (bf16 elems, multiple of 8)

// ---------------- WMMA fragment loaders ----------------

// A matrix 16x32 bf16 (contiguous k order used here, not the ISA table order): lane L (m = L&15, hl = L>>4);
// VGPRs 0..3 hold K = hl*16 .. hl*16+7   (contiguous -> one b128)
// VGPRs 4..7 hold K = hl*16+8 .. +7      (contiguous -> one b128)
__device__ inline v16bf frag_a_lds(const __bf16* S, int stride, int m0, int kk, int lane) {
  const int m  = m0 + (lane & 15);
  const int hl = lane >> 4;
  const __bf16* base = S + m * stride + kk * 32 + hl * 16;
  union { v16bf v; uint4 q[2]; } u;
  u.q[0] = *(const uint4*)(base);        // ds_load_b128
  u.q[1] = *(const uint4*)(base + 8);    // ds_load_b128
  return u.v;
}

// B fragment from pre-packed weights: 32 bytes contiguous per lane.
// Packed layout: dst[(((j*NK + kk)*32 + lane)*16 + e] = W[kk*32 + (lane>>4)*16 + e][j*16 + (lane&15)]
__device__ inline v16bf frag_b_pk(const __bf16* __restrict__ Wpk, int NK, int j, int kk, int lane) {
  const __bf16* p = Wpk + ((((size_t)j * NK + kk) * 32 + lane) << 4);
  union { v16bf v; uint4 q[2]; } u;
  u.q[0] = *(const uint4*)(p);           // global_load_b128
  u.q[1] = *(const uint4*)(p + 8);       // global_load_b128
  return u.v;
}

// ---------------- small utility kernels ----------------

__global__ void zero_kernel(float* __restrict__ p, long n) {
  long g = (long)blockIdx.x * blockDim.x + threadIdx.x;
  if (g < n) p[g] = 0.f;
}

// Pack W (L x K x C, row-major f32) into bf16 WMMA-B-fragment-major order.
__global__ void pack_b_kernel(const float* __restrict__ W, __bf16* __restrict__ dst,
                              int L, int K, int C) {
  long idx = (long)blockIdx.x * blockDim.x + threadIdx.x;
  const int NK = K / 32, NJ = C / 16;
  long total = (long)L * NJ * NK * 512;
  if (idx >= total) return;
  int e    = (int)(idx & 15);
  int lane = (int)(idx >> 4) & 31;
  long t   = idx >> 9;
  int kk = (int)(t % NK); t /= NK;
  int j  = (int)(t % NJ); t /= NJ;
  int l  = (int)t;
  int k   = kk * 32 + (lane >> 4) * 16 + e;
  int col = j * 16 + (lane & 15);
  dst[idx] = (__bf16)W[((size_t)l * K + k) * C + col];
}

__global__ void copy4_kernel(const float* __restrict__ src, float* __restrict__ dst, long n4) {
  long g = (long)blockIdx.x * blockDim.x + threadIdx.x;
  if (g < n4) ((float4*)dst)[g] = ((const float4*)src)[g];
}

// h[node, :] = emb[x[node*stride], :]   (wave-per-node, float4 per lane)
__global__ void embed_kernel(const int* __restrict__ x, const float* __restrict__ emb,
                             float* __restrict__ h, int N, int xstride) {
  long g = (long)blockIdx.x * blockDim.x + threadIdx.x;
  int node = (int)(g >> 5), lane = (int)(g & 31);
  if (node >= N) return;
  int id = x[(size_t)node * xstride];
  int c  = lane * 4;
  float4 v = *(const float4*)(emb + (size_t)id * D + c);
  *(float4*)(h + (size_t)node * D + c) = v;
}

// msg = relu(h[src] + bond_emb[type]); z[dst] += msg   (wave-per-edge)
__global__ void edge_scatter_kernel(const float* __restrict__ h, const int* __restrict__ ei,
                                    const int* __restrict__ ea, const float* __restrict__ bemb,
                                    float* __restrict__ z, int E) {
  long g = (long)blockIdx.x * blockDim.x + threadIdx.x;
  int edge = (int)(g >> 5), lane = (int)(g & 31);
  if (edge >= E) return;
  int src = ei[edge];
  int dst = ei[(size_t)E + edge];
  int bt  = ea[(size_t)edge * 3];
  int c   = lane * 4;
  float4 hv = *(const float4*)(h + (size_t)src * D + c);
  float4 ev = *(const float4*)(bemb + (size_t)bt * D + c);
  float m0 = fmaxf(hv.x + ev.x, 0.f);
  float m1 = fmaxf(hv.y + ev.y, 0.f);
  float m2 = fmaxf(hv.z + ev.z, 0.f);
  float m3 = fmaxf(hv.w + ev.w, 0.f);
  float* zp = z + (size_t)dst * D + c;
  atomicAdd(zp + 0, m0);
  atomicAdd(zp + 1, m1);
  atomicAdd(zp + 2, m2);
  atomicAdd(zp + 3, m3);
}

// ---------------- fused MLP (two bf16 WMMA GEMMs) + BN partial sums ----------------

__global__ __launch_bounds__(128) void mlp_bn_stats_kernel(
    const float* __restrict__ z,
    const __bf16* __restrict__ W1pk, const float* __restrict__ b1,
    const __bf16* __restrict__ W2pk, const float* __restrict__ b2,
    float* __restrict__ z2, float* __restrict__ gsum, float* __restrict__ gsq, int N) {
  __shared__ union TileU {
    __bf16 A[ROWS * AS_STRIDE];   // input tile (phase 1)
    __bf16 Y[ROWS * YS_STRIDE];   // hidden tile (phase 2), aliased
  } T;
  __shared__ float sSum[D], sSq[D];

  const int tid  = threadIdx.x;
  const int lane = tid & 31;
  const int wave = tid >> 5;
  const int rowBase = blockIdx.x * ROWS;

  if (tid < D) { sSum[tid] = 0.f; sSq[tid] = 0.f; }
  if (tid == 0) { __builtin_prefetch(W1pk, 0, 1); __builtin_prefetch(W2pk, 0, 1); }

  // stage input tile to LDS as bf16 (zero-pad past N)
  for (int i = tid; i < ROWS * (D / 4); i += blockDim.x) {
    int r = i / (D / 4);
    int c = (i % (D / 4)) * 4;
    int row = rowBase + r;
    float4 v = make_float4(0.f, 0.f, 0.f, 0.f);
    if (row < N) v = *(const float4*)(z + (size_t)row * D + c);
    T.A[r * AS_STRIDE + c + 0] = (__bf16)v.x;
    T.A[r * AS_STRIDE + c + 1] = (__bf16)v.y;
    T.A[r * AS_STRIDE + c + 2] = (__bf16)v.z;
    T.A[r * AS_STRIDE + c + 3] = (__bf16)v.w;
  }
  __syncthreads();

  const int m0 = wave * 16;
  const int n  = lane & 15;
  const int hl = lane >> 4;

  // GEMM1: [16 x 128] x [128 x 256] -> relu -> bf16 hidden tile in LDS
  v16bf a1[4];
#pragma unroll
  for (int kk = 0; kk < 4; ++kk) a1[kk] = frag_a_lds(T.A, AS_STRIDE, m0, kk, lane);
  __syncthreads();  // everyone done reading A before Y aliases it

  for (int j = 0; j < 16; ++j) {
    const float bias = b1[j * 16 + n];
    v8f c;
#pragma unroll
    for (int r = 0; r < 8; ++r) c[r] = bias;
#pragma unroll
    for (int kk = 0; kk < 4; ++kk) {
      v16bf b = frag_b_pk(W1pk, 4, j, kk, lane);
      c = __builtin_amdgcn_wmma_f32_16x16x32_bf16(false, a1[kk], false, b,
                                                  (short)0, c, false, false);
    }
#pragma unroll
    for (int r = 0; r < 8; ++r) {
      float y = c[r] > 0.f ? c[r] : 0.f;
      T.Y[(m0 + r + 8 * hl) * YS_STRIDE + j * 16 + n] = (__bf16)y;  // D layout: row r + 8*hl
    }
  }
  __syncthreads();

  // GEMM2: [16 x 256] x [256 x 128] + b2 -> z2, with fused BN partial sums
  v16bf a2[8];
#pragma unroll
  for (int kk = 0; kk < 8; ++kk) a2[kk] = frag_a_lds(T.Y, YS_STRIDE, m0, kk, lane);

  for (int j = 0; j < 8; ++j) {
    const int cb = j * 16;
    const float bias = b2[cb + n];
    v8f c;
#pragma unroll
    for (int r = 0; r < 8; ++r) c[r] = bias;
#pragma unroll
    for (int kk = 0; kk < 8; ++kk) {
      v16bf b = frag_b_pk(W2pk, 8, j, kk, lane);
      c = __builtin_amdgcn_wmma_f32_16x16x32_bf16(false, a2[kk], false, b,
                                                  (short)0, c, false, false);
    }
    float ps = 0.f, pq = 0.f;
#pragma unroll
    for (int r = 0; r < 8; ++r) {
      int row = rowBase + m0 + r + 8 * hl;
      if (row < N) {
        z2[(size_t)row * D + cb + n] = c[r];
        ps += c[r];
        pq += c[r] * c[r];
      }
    }
    atomicAdd(&sSum[cb + n], ps);  // ds_add_f32
    atomicAdd(&sSq[cb + n], pq);
  }
  __syncthreads();
  if (tid < D) {
    atomicAdd(&gsum[tid], sSum[tid]);
    atomicAdd(&gsq[tid], sSq[tid]);
  }
}

// ---------------- BN finalize / apply ----------------

__global__ void bn_finalize_kernel(const float* __restrict__ gsum, const float* __restrict__ gsq,
                                   const float* __restrict__ gamma, const float* __restrict__ beta,
                                   float* __restrict__ scale, float* __restrict__ shift, int N) {
  int c = threadIdx.x;
  float inv  = 1.f / (float)N;
  float mean = gsum[c] * inv;
  float var  = gsq[c] * inv - mean * mean;
  float sc   = gamma[c] * rsqrtf(var + 1e-5f);
  scale[c] = sc;
  shift[c] = beta[c] - mean * sc;
}

__global__ void bn_apply_kernel(const float* __restrict__ z2, const float* __restrict__ scale,
                                const float* __restrict__ shift, float* __restrict__ h,
                                int N, int do_relu) {
  long g = (long)blockIdx.x * blockDim.x + threadIdx.x;
  int node = (int)(g >> 5), lane = (int)(g & 31);
  if (node >= N) return;
  int c = lane * 4;
  float4 v = *(const float4*)(z2 + (size_t)node * D + c);
  float4 s = *(const float4*)(scale + c);
  float4 t = *(const float4*)(shift + c);
  float4 o;
  o.x = v.x * s.x + t.x;
  o.y = v.y * s.y + t.y;
  o.z = v.z * s.z + t.z;
  o.w = v.w * s.w + t.w;
  if (do_relu) {
    o.x = fmaxf(o.x, 0.f); o.y = fmaxf(o.y, 0.f);
    o.z = fmaxf(o.z, 0.f); o.w = fmaxf(o.w, 0.f);
  }
  *(float4*)(h + (size_t)node * D + c) = o;
}

// out[batch[node]] += h[node]   (wave-per-node)
__global__ void pool_kernel(const float* __restrict__ h, const int* __restrict__ batch,
                            float* __restrict__ out, int N) {
  long g = (long)blockIdx.x * blockDim.x + threadIdx.x;
  int node = (int)(g >> 5), lane = (int)(g & 31);
  if (node >= N) return;
  int grp = batch[node];
  int c = lane * 4;
  float4 v = *(const float4*)(h + (size_t)node * D + c);
  float* o = out + (size_t)grp * D + c;
  atomicAdd(o + 0, v.x);
  atomicAdd(o + 1, v.y);
  atomicAdd(o + 2, v.z);
  atomicAdd(o + 3, v.w);
}

// ---------------- host launcher ----------------

extern "C" void kernel_launch(void* const* d_in, const int* in_sizes, int n_in,
                              void* d_out, int out_size, void* d_ws, size_t ws_size,
                              hipStream_t stream) {
  const int*   x     = (const int*)d_in[0];
  const int*   ei    = (const int*)d_in[1];
  const int*   ea    = (const int*)d_in[2];
  const int*   batch = (const int*)d_in[3];
  const float* aemb  = (const float*)d_in[4];
  const float* bemb  = (const float*)d_in[5];
  const float* W1    = (const float*)d_in[6];
  const float* b1    = (const float*)d_in[7];
  const float* W2    = (const float*)d_in[8];
  const float* b2    = (const float*)d_in[9];
  const float* gamma = (const float*)d_in[10];
  const float* beta  = (const float*)d_in[11];
  float* out = (float*)d_out;

  const int N = in_sizes[0] / 9;
  const int E = in_sizes[1] / 2;

  // workspace carve
  char* ws = (char*)d_ws;
  float* h  = (float*)ws;  ws += (size_t)N * D * sizeof(float);
  float* zb = (float*)ws;  ws += (size_t)N * D * sizeof(float);
  float* z2 = (float*)ws;  ws += (size_t)N * D * sizeof(float);
  __bf16* W1pk = (__bf16*)ws;  ws += (size_t)LAYERS * D * 2 * D * sizeof(__bf16);
  __bf16* W2pk = (__bf16*)ws;  ws += (size_t)LAYERS * 2 * D * D * sizeof(__bf16);
  float* gsum  = (float*)ws;  ws += D * sizeof(float);
  float* gsq   = (float*)ws;  ws += D * sizeof(float);
  float* scale = (float*)ws;  ws += D * sizeof(float);
  float* shift = (float*)ws;  ws += D * sizeof(float);

  const long nw = (long)LAYERS * D * 2 * D;  // elems per weight tensor family
  pack_b_kernel<<<(int)((nw + 255) / 256), 256, 0, stream>>>(W1, W1pk, LAYERS, D, 2 * D);
  pack_b_kernel<<<(int)((nw + 255) / 256), 256, 0, stream>>>(W2, W2pk, LAYERS, 2 * D, D);

  const int nodeBlocks = (int)(((long)N * 32 + 255) / 256);
  const int edgeBlocks = (int)(((long)E * 32 + 255) / 256);
  const long n4 = (long)N * D / 4;
  const int copyBlocks = (int)((n4 + 255) / 256);
  const int mlpBlocks  = (N + ROWS - 1) / ROWS;

  embed_kernel<<<nodeBlocks, 256, 0, stream>>>(x, aemb, h, N, 9);
  zero_kernel<<<(int)(((long)out_size + 255) / 256), 256, 0, stream>>>(out, (long)out_size);

  const long perW1 = (long)2 * D * 2 * D;   // 2*D*2*D = 65536 elems per-layer stride
  const long perW2 = (long)2 * D * 2 * D;   // 2*D*2*D = 65536 elems per-layer stride

  for (int l = 0; l < LAYERS; ++l) {
    copy4_kernel<<<copyBlocks, 256, 0, stream>>>(h, zb, n4);                       // z = h
    edge_scatter_kernel<<<edgeBlocks, 256, 0, stream>>>(h, ei, ea, bemb, zb, E);   // z += agg
    zero_kernel<<<1, 256, 0, stream>>>(gsum, 2 * D);  // gsum & gsq are adjacent
    mlp_bn_stats_kernel<<<mlpBlocks, 128, 0, stream>>>(
        zb, W1pk + (size_t)l * perW1, b1 + (size_t)l * 2 * D,
        W2pk + (size_t)l * perW2, b2 + (size_t)l * D,
        z2, gsum, gsq, N);
    bn_finalize_kernel<<<1, D, 0, stream>>>(gsum, gsq, gamma + (size_t)l * D,
                                            beta + (size_t)l * D, scale, shift, N);
    bn_apply_kernel<<<nodeBlocks, 256, 0, stream>>>(z2, scale, shift, h, N,
                                                    l < LAYERS - 1 ? 1 : 0);
  }

  pool_kernel<<<nodeBlocks, 256, 0, stream>>>(h, batch, out, N);
}
